// WindowAttention_kv_42674795053150
// MI455X (gfx1250) — matrix operand order repaired and checked by reading, not compiled, not run on hardware
//
#include <hip/hip_runtime.h>
#include <hip/hip_bf16.h>

typedef __attribute__((ext_vector_type(16))) _Float16 v16h;
typedef __attribute__((ext_vector_type(8)))  _Float16 v8h;
typedef __attribute__((ext_vector_type(4)))  _Float16 v4h;
typedef __attribute__((ext_vector_type(8)))  float    v8f;

#define NTOK 343           // 7*7*7
#define NPAD 352           // 22*16
#define CDIM 192
#define HEADS 6
#define HD 32
#define BWIN 256
#define NW 32
#define ROWS (BWIN * NTOK) // 87808
#define KT 22              // NPAD/16 key tiles for QK
#define KT2 11             // NPAD/32 key chunks for PV

static __device__ __forceinline__ v16h cat8(v8h lo, v8h hi) {
    return __builtin_shufflevector(lo, hi, 0,1,2,3,4,5,6,7,8,9,10,11,12,13,14,15);
}

static __device__ __forceinline__ v8f wmma_f16(v16h a, v16h b, v8f c) {
    // D = A(16x32 f16) x B(32x16 f16) + C(16x16 f32)
    return __builtin_amdgcn_wmma_f32_16x16x32_f16(false, a, false, b, (short)0, c, false, false);
}

// ---------------- prep kernels ----------------

__global__ void prep_q(const float* __restrict__ x, _Float16* __restrict__ q, int n) {
    int i = blockIdx.x * 256 + threadIdx.x;
    if (i < n) q[i] = (_Float16)(x[i] * 0.17677669529663687f); // * D^-0.5
}

__global__ void prep_wT(const float* __restrict__ kvw, const float* __restrict__ pw,
                        _Float16* __restrict__ kvwT, _Float16* __restrict__ pwT) {
    int i = blockIdx.x * 256 + threadIdx.x;
    if (i < 384 * CDIM) { int n = i / CDIM, k = i % CDIM; kvwT[i] = (_Float16)kvw[k * 384 + n]; }
    if (i < CDIM * CDIM) { int n = i / CDIM, k = i % CDIM; pwT[i]  = (_Float16)pw[k * CDIM + n]; }
}

__global__ void prep_bias(const int* __restrict__ rel, const float* __restrict__ table,
                          float* __restrict__ biasf) {
    int i = blockIdx.x * 256 + threadIdx.x;
    const int NN = NTOK * NTOK;
    if (i < HEADS * NN) {
        int h = i / NN, ij = i % NN;
        biasf[i] = table[rel[ij] * HEADS + h];
    }
}

__global__ void zero_vpad(_Float16* __restrict__ vT) {
    int i = blockIdx.x * 256 + threadIdx.x;
    const int PAD = NPAD - NTOK; // 9
    if (i < BWIN * HEADS * HD * PAD) {
        int bhd = i / PAD, p = i % PAD;
        vT[(size_t)bhd * NPAD + NTOK + p] = (_Float16)0.f;
    }
}

// ---------------- KV projection GEMM (A staged in LDS; 8 waves x 3 n-tiles) ----------------

__global__ __launch_bounds__(256, 1)
void kv_gemm(const float* __restrict__ skip, const _Float16* __restrict__ kvwT,
             const float* __restrict__ kvb, _Float16* __restrict__ kf16,
             _Float16* __restrict__ vT) {
    __shared__ alignas(16) _Float16 As[16][CDIM];   // 6 KB
    const int tid = threadIdx.x;
    const int lane = tid & 31;
    const int wave = tid >> 5;
    const int mt = blockIdx.x;                      // 0..5487

    // Stage A strip (16 x 192 f32 -> f16), read from HBM exactly once per row-tile.
    for (int c = tid; c < 16 * CDIM / 4; c += 256) { // 768 float4 chunks
        int row = c / (CDIM / 4), k4 = (c % (CDIM / 4)) * 4;
        float4 a = *(const float4*)(skip + (size_t)(mt * 16 + row) * CDIM + k4);
        v4h hv; hv[0]=(_Float16)a.x; hv[1]=(_Float16)a.y; hv[2]=(_Float16)a.z; hv[3]=(_Float16)a.w;
        *(v4h*)&As[row][k4] = hv;
    }
    __syncthreads();

    const int nloc = lane & 15, half = lane >> 4;
    v8f acc[3] = {};
    #pragma unroll
    for (int k0 = 0; k0 < CDIM; k0 += 32) {
        const _Float16* ap = &As[nloc][k0 + half * 8];
        v16h a = cat8(*(const v8h*)ap, *(const v8h*)(ap + 16));
        #pragma unroll
        for (int j = 0; j < 3; ++j) {
            int nt = wave * 3 + j;                   // 0..23
            const _Float16* bp = kvwT + (size_t)(nt * 16 + nloc) * CDIM + k0 + half * 8;
            acc[j] = wmma_f16(a, cat8(*(const v8h*)bp, *(const v8h*)(bp + 16)), acc[j]);
        }
    }
    #pragma unroll
    for (int j = 0; j < 3; ++j) {
        const int nout = (wave * 3 + j) * 16 + nloc;
        const float bias = kvb[nout];
        #pragma unroll
        for (int r = 0; r < 8; ++r) {
            int orow = mt * 16 + half * 8 + r;
            float val = acc[j][r] + bias;
            if (nout < CDIM) {
                kf16[(size_t)orow * CDIM + nout] = (_Float16)val;
            } else {
                int c = nout - CDIM;
                int hh = c >> 5, d = c & 31;
                int bb = orow / NTOK, key = orow - bb * NTOK;
                vT[(size_t)((bb * HEADS + hh) * HD + d) * NPAD + key] = (_Float16)val;
            }
        }
    }
}

// ---------------- fused window attention (K/V slices staged in LDS) ----------------

__global__ __launch_bounds__(128, 1)
void attn_fused(const _Float16* __restrict__ qf16, const _Float16* __restrict__ kf16,
                const _Float16* __restrict__ vT, const float* __restrict__ biasf,
                const float* __restrict__ mask, _Float16* __restrict__ xf16) {
    __shared__ alignas(16) _Float16 probs[4][16][NPAD]; // 45 KB
    __shared__ alignas(16) _Float16 ksl[NPAD][HD];      // 22 KB, [key][d]
    __shared__ alignas(16) _Float16 vsl[HD][NPAD];      // 22 KB, [d][key]
    const int tid = threadIdx.x;
    const int lane = tid & 31;
    const int wave = tid >> 5;
    const int b = blockIdx.z;
    const int h = blockIdx.y;
    const int qbase = blockIdx.x * 64 + wave * 16;      // may exceed NTOK; no early exit
    const int nloc = lane & 15;
    const int half = lane >> 4;

    // Stage K slice (zero-padded) and V slice for this (b,h), shared by all 4 waves.
    for (int c = tid; c < NPAD * HD / 8; c += 128) {    // 1408 v8h chunks
        int key = c >> 2, part = (c & 3) * 8;
        v8h x = {};
        if (key < NTOK)
            x = *(const v8h*)(kf16 + (size_t)(b * NTOK + key) * CDIM + h * HD + part);
        *(v8h*)&ksl[key][part] = x;
    }
    for (int c = tid; c < HD * NPAD / 8; c += 128) {    // 1408 v8h chunks
        int d = c / (NPAD / 8), p = (c % (NPAD / 8)) * 8;
        *(v8h*)&vsl[d][p] = *(const v8h*)(vT + (size_t)((b * HEADS + h) * HD + d) * NPAD + p);
    }
    __syncthreads();

    // Q A-tile (persistent across all key tiles)
    int qrow = qbase + nloc; if (qrow > NTOK - 1) qrow = NTOK - 1;
    const _Float16* qp = qf16 + ((size_t)(b * NTOK + qrow)) * CDIM + h * HD + half * 8;
    v16h qa = cat8(*(const v8h*)qp, *(const v8h*)(qp + 16));

    const int w = b & (NW - 1);
    const float* bias_h = biasf + (size_t)h * NTOK * NTOK;
    const float* mask_w = mask + (size_t)w * NTOK * NTOK;

    // Phase 1: full logit strip (16 x 352) in registers
    v8f acc[KT];
    #pragma unroll
    for (int kt = 0; kt < KT; ++kt) {
        int key = kt * 16 + nloc;
        const _Float16* kp = &ksl[kt * 16 + nloc][half * 8];
        v16h kb = cat8(*(const v8h*)kp, *(const v8h*)(kp + 16));
        v8f c0 = {};
        acc[kt] = wmma_f16(qa, kb, c0);
        #pragma unroll
        for (int r = 0; r < 8; ++r) {
            int row = qbase + half * 8 + r;
            float val;
            if (key <= NTOK - 1 && row <= NTOK - 1)
                val = acc[kt][r] + bias_h[row * NTOK + key] + mask_w[row * NTOK + key];
            else
                val = -1e30f;
            acc[kt][r] = val;
        }
    }

    // Phase 2: softmax (row stats via 16-lane xor reductions)
    float rmax[8], rsum[8];
    #pragma unroll
    for (int r = 0; r < 8; ++r) {
        float m = -1e30f;
        #pragma unroll
        for (int kt = 0; kt < KT; ++kt) m = fmaxf(m, acc[kt][r]);
        #pragma unroll
        for (int s = 1; s < 16; s <<= 1) m = fmaxf(m, __shfl_xor(m, s, 16));
        rmax[r] = m;
    }
    #pragma unroll
    for (int r = 0; r < 8; ++r) {
        float ssum = 0.f;
        #pragma unroll
        for (int kt = 0; kt < KT; ++kt) {
            float e = __expf(acc[kt][r] - rmax[r]);
            acc[kt][r] = e;
            ssum += e;
        }
        #pragma unroll
        for (int s = 1; s < 16; s <<= 1) ssum += __shfl_xor(ssum, s, 16);
        rsum[r] = ssum;
    }

    // Phase 3: normalized probs -> LDS (f16, [row][key], wave-private)
    #pragma unroll
    for (int r = 0; r < 8; ++r) {
        float inv = 1.f / rsum[r];
        int prow = half * 8 + r;
        #pragma unroll
        for (int kt = 0; kt < KT; ++kt)
            probs[wave][prow][kt * 16 + nloc] = (_Float16)(acc[kt][r] * inv);
    }
    asm volatile("s_wait_dscnt 0" ::: "memory"); // same-wave LDS RAW

    // Phase 4: P @ V  (2 d-tiles of 16, K chunks of 32 keys)
    #pragma unroll
    for (int dt = 0; dt < 2; ++dt) {
        v8f o = {};
        #pragma unroll
        for (int c2 = 0; c2 < KT2; ++c2) {
            int kb0 = c2 * 32;
            const _Float16* pp = &probs[wave][nloc][kb0 + half * 8];
            v16h pa = cat8(*(const v8h*)pp, *(const v8h*)(pp + 16));
            const _Float16* vp = &vsl[dt * 16 + nloc][kb0 + half * 8];
            v16h vb = cat8(*(const v8h*)vp, *(const v8h*)(vp + 16));
            o = wmma_f16(pa, vb, o);
        }
        #pragma unroll
        for (int r = 0; r < 8; ++r) {
            int row = qbase + half * 8 + r;
            if (row <= NTOK - 1)
                xf16[((size_t)(b * NTOK + row)) * CDIM + h * HD + dt * 16 + nloc] = (_Float16)o[r];
        }
    }
}

// ---------------- output projection GEMM (A staged in LDS; 4 waves x 3 n-tiles) ----------------

__global__ __launch_bounds__(128, 1)
void proj_gemm(const _Float16* __restrict__ xf16, const _Float16* __restrict__ pwT,
               const float* __restrict__ pb, float* __restrict__ out) {
    __shared__ alignas(16) _Float16 As[16][CDIM];   // 6 KB
    const int tid = threadIdx.x;
    const int lane = tid & 31;
    const int wave = tid >> 5;
    const int mt = blockIdx.x;                      // 0..5487

    for (int c = tid; c < 16 * CDIM / 8; c += 128) { // 384 v8h chunks
        int row = c / (CDIM / 8), k8 = (c % (CDIM / 8)) * 8;
        *(v8h*)&As[row][k8] = *(const v8h*)(xf16 + (size_t)(mt * 16 + row) * CDIM + k8);
    }
    __syncthreads();

    const int nloc = lane & 15, half = lane >> 4;
    v8f acc[3] = {};
    #pragma unroll
    for (int k0 = 0; k0 < CDIM; k0 += 32) {
        const _Float16* ap = &As[nloc][k0 + half * 8];
        v16h a = cat8(*(const v8h*)ap, *(const v8h*)(ap + 16));
        #pragma unroll
        for (int j = 0; j < 3; ++j) {
            int nt = wave * 3 + j;                   // 0..11
            const _Float16* bp = pwT + (size_t)(nt * 16 + nloc) * CDIM + k0 + half * 8;
            acc[j] = wmma_f16(a, cat8(*(const v8h*)bp, *(const v8h*)(bp + 16)), acc[j]);
        }
    }
    #pragma unroll
    for (int j = 0; j < 3; ++j) {
        const int nout = (wave * 3 + j) * 16 + nloc;
        const float bias = pb[nout];
        #pragma unroll
        for (int r = 0; r < 8; ++r) {
            int orow = mt * 16 + half * 8 + r;
            out[(size_t)orow * CDIM + nout] = acc[j][r] + bias;
        }
    }
}

// ---------------- launcher ----------------

extern "C" void kernel_launch(void* const* d_in, const int* in_sizes, int n_in,
                              void* d_out, int out_size, void* d_ws, size_t ws_size,
                              hipStream_t stream) {
    (void)in_sizes; (void)n_in; (void)out_size; (void)ws_size;
    const float* skip  = (const float*)d_in[0];
    const float* x_up  = (const float*)d_in[1];
    const float* mask  = (const float*)d_in[2];
    const int*   rel   = (const int*)d_in[3];
    const float* table = (const float*)d_in[4];
    const float* kv_w  = (const float*)d_in[5];
    const float* kv_b  = (const float*)d_in[6];
    const float* pw    = (const float*)d_in[7];
    const float* pb    = (const float*)d_in[8];
    float* out = (float*)d_out;

    char* ws = (char*)d_ws;
    size_t o = 0;
    _Float16* qf16 = (_Float16*)(ws + o); o += (size_t)ROWS * CDIM * 2;              // 33.7 MB
    _Float16* kf16 = (_Float16*)(ws + o); o += (size_t)ROWS * CDIM * 2;              // 33.7 MB
    _Float16* vT   = (_Float16*)(ws + o); o += (size_t)BWIN * HEADS * HD * NPAD * 2; // 34.6 MB
    _Float16* xf16 = (_Float16*)(ws + o); o += (size_t)ROWS * CDIM * 2;              // 33.7 MB
    _Float16* kvwT = (_Float16*)(ws + o); o += (size_t)384 * CDIM * 2;
    _Float16* pwT  = (_Float16*)(ws + o); o += (size_t)CDIM * CDIM * 2;
    float*    biasf= (float*)(ws + o);    o += (size_t)HEADS * NTOK * NTOK * 4;      // 2.8 MB

    // prep
    {
        int n = ROWS * CDIM;
        prep_q<<<(n + 255) / 256, 256, 0, stream>>>(x_up, qf16, n);
        prep_wT<<<(384 * CDIM + 255) / 256, 256, 0, stream>>>(kv_w, pw, kvwT, pwT);
        prep_bias<<<(HEADS * NTOK * NTOK + 255) / 256, 256, 0, stream>>>(rel, table, biasf);
        zero_vpad<<<(BWIN * HEADS * HD * (NPAD - NTOK) + 255) / 256, 256, 0, stream>>>(vT);
    }
    // kv projection: M=87808, N=384, K=192
    kv_gemm<<<dim3(ROWS / 16), 256, 0, stream>>>(skip, kvwT, kv_b, kf16, vT);
    // fused attention
    attn_fused<<<dim3((NTOK + 63) / 64, HEADS, BWIN), 128, 0, stream>>>(qf16, kf16, vT, biasf, mask, xf16);
    // output projection: M=87808, N=192, K=192
    proj_gemm<<<dim3(ROWS / 16), 128, 0, stream>>>(xf16, pwT, pb, out);
}
